// CosRec_base_2259152798115
// MI455X (gfx1250) — matrix operand order repaired and checked by reading, not compiled, not run on hardware
//
#include <hip/hip_runtime.h>
#include <hip/hip_bf16.h>

typedef __attribute__((ext_vector_type(16))) _Float16 v16h;
typedef __attribute__((ext_vector_type(8)))  float    v8f;

#define B_    512
#define L_    32
#define D_    64
#define F_    100
#define T_    3
#define FPAD  132   // padded f (K) row stride in floats: 528B -> bank advance 4/row, b64-aligned
#define GPAD  112   // padded g (N) dimension: 7 tiles of 16
#define NT_   7
#define KC_   4
#define MT_   64    // M tiles: 1024 rows / 16
#define W2DIM 164

// LDS layout (bytes), total 63360 < 64KB:
//   [0     .. 16896)  ApB : float[32][132]  (A + b1, f-padded with 0)
//   [16896 .. 33792)  Bm  : float[32][132]
//   [33792 .. 62464)  wf2p: _Float16[7*4*32*16]  (B-operand-packed Wf2)
//   [33792 .. 41984)  emb : float[32][64]   (ALIAS; dead before wf2p written)
//   [62464 .. 62912)  bf2s: float[112]
//   [62912 .. 63360)  xs  : float[112]
#define SMEM_BYTES 63360

__global__ __launch_bounds__(256)
void cosrec_fused_kernel(const int* __restrict__ seq_var,
                         const int* __restrict__ user_var,
                         const int* __restrict__ item_var,
                         const float* __restrict__ item_emb,
                         const float* __restrict__ user_emb,
                         const float* __restrict__ W2,
                         const float* __restrict__ b2,
                         const float* __restrict__ W1,
                         const float* __restrict__ b1,
                         const float* __restrict__ Wf2,
                         const float* __restrict__ bf2,
                         float* __restrict__ out)
{
    __shared__ __align__(64) unsigned char smem_raw[SMEM_BYTES];
    float*     ApB  = (float*)(smem_raw);
    float*     Bm   = (float*)(smem_raw + 16896);
    _Float16*  wf2p = (_Float16*)(smem_raw + 33792);
    float*     emb  = (float*)(smem_raw + 33792);   // alias (stages 0-1a only)
    float*     bf2s = (float*)(smem_raw + 62464);
    float*     xs   = (float*)(smem_raw + 62912);

    const int b     = blockIdx.x;
    const int tid   = threadIdx.x;
    const int lane  = tid & 31;
    const int wave  = tid >> 5;         // 0..7
    const int lmod  = lane & 15;
    const int lhalf = lane >> 4;        // 0/1

    // ---- Stage 0: gather sequence embeddings; init xs / bf2s -------------
    for (int i = tid; i < L_ * D_; i += 256) {
        int l = i >> 6, k = i & 63;
        long idx = seq_var[b * L_ + l];
        emb[i] = item_emb[idx * D_ + k];
    }
    for (int i = tid; i < GPAD; i += 256) {
        xs[i]   = 0.0f;
        bf2s[i] = (i < F_) ? bf2[i] : 0.0f;
    }
    // Prefetch the stage-3 W2 rows; latency hidden behind the whole GEMM phase.
    if (tid < T_) {
        long it = item_var[b * T_ + tid];
        __builtin_prefetch(&W2[it * W2DIM], 0, 0);
    }
    __syncthreads();

    // ---- Stage 1a: first MLP layer (tiny) --------------------------------
    // ApB[l][f] = dot(emb[l], W1[f][0:64]) + b1[f] ; Bm[l][f] = dot(emb[l], W1[f][64:128])
    for (int i = tid; i < L_ * FPAD; i += 256) {
        int l = i / FPAD, f = i - l * FPAD;
        float sA = 0.f, sB = 0.f;
        if (f < F_) {
            sA = b1[f];
            const float* w = &W1[f * (2 * D_)];
            const float* e = &emb[l * D_];
            #pragma unroll 8
            for (int k = 0; k < D_; ++k) {
                float ev = e[k];
                sA += ev * w[k];
                sB += ev * w[D_ + k];
            }
        }
        ApB[l * FPAD + f] = sA;
        Bm [l * FPAD + f] = sB;
    }
    __syncthreads();

    // ---- Stage 1b: pack Wf2 as f16 B-operand tiles (emb now dead) --------
    // wf2p[(((nt*4)+kc)*32 + ln)*16 + h] = Wf2[g][f] with
    //   g = nt*16 + (ln&15), f = kc*32 + (ln>>4)*16 + h   (zero padding)
    for (int i = tid; i < NT_ * KC_ * 32 * 16; i += 256) {
        int h  = i & 15;
        int ln = (i >> 4) & 31;
        int kc = (i >> 9) & 3;
        int nt = i >> 11;
        int g = nt * 16 + (ln & 15);
        int f = kc * 32 + (ln >> 4) * 16 + h;
        float v = (g < F_ && f < F_) ? Wf2[g * F_ + f] : 0.0f;
        wf2p[i] = (_Float16)v;
    }
    __syncthreads();

    // ---- Stage 2: fused H formation + 100x100 GEMM + relu + row-sum ------
    float colacc[NT_];
    #pragma unroll
    for (int t = 0; t < NT_; ++t) colacc[t] = 0.f;

    for (int mt = wave; mt < MT_; mt += 8) {
        int row = mt * 16 + lmod;       // this lane's A-operand row (M = lane&15)
        int ai  = row >> 5;             // a index
        int ci  = row & 31;             // c index
        const float* ap = &ApB[ci * FPAD];
        const float* bp = &Bm [ai * FPAD];

        v8f acc[NT_];
        #pragma unroll
        for (int nt = 0; nt < NT_; ++nt) acc[nt] = (v8f){};

        // kc outer / nt inner: one A operand feeds 7 consecutive WMMAs, and
        // consecutive WMMAs hit different accumulators (no D->C RAW stalls).
        #pragma unroll
        for (int kc = 0; kc < KC_; ++kc) {
            v16h aop;
            #pragma unroll
            for (int h = 0; h < 16; ++h) {
                int f = kc * 32 + (lhalf << 4) + h;
                aop[h] = (_Float16)fmaxf(ap[f] + bp[f], 0.f);
            }
            #pragma unroll
            for (int nt = 0; nt < NT_; ++nt) {
                const v16h bop = *(const v16h*)&wf2p[(((nt * KC_) + kc) * 32 + lane) * 16];
                acc[nt] = __builtin_amdgcn_wmma_f32_16x16x32_f16(
                              false, aop, false, bop, (short)0, acc[nt], false, false);
            }
        }

        // epilogue: + bf2, relu, sum this lane's 8 rows of each D tile
        #pragma unroll
        for (int nt = 0; nt < NT_; ++nt) {
            float bb = bf2s[nt * 16 + lmod];
            float s = 0.f;
            #pragma unroll
            for (int j = 0; j < 8; ++j)
                s += fmaxf(acc[nt][j] + bb, 0.f);
            colacc[nt] += s;
        }
    }
    #pragma unroll
    for (int nt = 0; nt < NT_; ++nt)
        atomicAdd(&xs[nt * 16 + lmod], colacc[nt]);
    __syncthreads();

    // ---- Stage 3: final scoring (3 tiny 164-dim dots) --------------------
    if (tid < T_) {
        int t = tid;
        long it = item_var[b * T_ + t];
        const float* w = &W2[it * W2DIM];
        float s = b2[it];
        for (int j = 0; j < F_; ++j) s += w[j] * xs[j];
        long u = user_var[b];
        const float* ue = &user_emb[u * D_];
        for (int j = 0; j < D_; ++j) s += w[F_ + j] * ue[j];
        out[b * T_ + t] = s;
    }
}

extern "C" void kernel_launch(void* const* d_in, const int* in_sizes, int n_in,
                              void* d_out, int out_size, void* d_ws, size_t ws_size,
                              hipStream_t stream) {
    (void)in_sizes; (void)n_in; (void)d_ws; (void)ws_size; (void)out_size;
    const int*   seq_var  = (const int*)  d_in[0];
    const int*   user_var = (const int*)  d_in[1];
    const int*   item_var = (const int*)  d_in[2];
    const float* item_emb = (const float*)d_in[3];
    const float* user_emb = (const float*)d_in[4];
    const float* W2       = (const float*)d_in[5];
    const float* b2       = (const float*)d_in[6];
    const float* W1       = (const float*)d_in[7];
    const float* b1       = (const float*)d_in[8];
    const float* Wf2      = (const float*)d_in[9];
    const float* bf2      = (const float*)d_in[10];
    float* out = (float*)d_out;

    cosrec_fused_kernel<<<B_, 256, 0, stream>>>(
        seq_var, user_var, item_var, item_emb, user_emb,
        W2, b2, W1, b1, Wf2, bf2, out);
}
